// AttentionAvg_11312943858015
// MI455X (gfx1250) — matrix operand order repaired and checked by reading, compiled, not run on hardware
//
#include <hip/hip_runtime.h>
#include <hip/hip_bf16.h>

// ---------------------------------------------------------------------------
// AttentionAvg for MI455X (gfx1250, wave32, WMMA + TDM).
//   B=8, T=4096, D=768.
//   Phase 0: Xb = bf16(X), Wqb = bf16(Wq), Wkb = bf16(Wk)
//   Phase 1: Q = X Wq^T + bq, K = X Wk^T + bk   (bf16 WMMA, f32 accum)
//   Phase 2: per query row t: rowsum = sum_k exp(q.k/sqrt(D)) (masked),
//            diag = exp(q_t.k_t/sqrt(D))  -> wdiag = diag/rowsum
//            K tiles staged LDS via tensor_load_to_lds (double buffered).
//   Phase 3: out[b,d] = sum_t X[b,t,d] * wdiag[b,t]
// Workspace: Xb(50.3MB) Qb(50.3MB) Kb(50.3MB) Wqb(1.2MB) Wkb(1.2MB) wd(128KB)
// ---------------------------------------------------------------------------

typedef __bf16 bf16_t;
typedef __attribute__((ext_vector_type(16))) __bf16 v16bf;
typedef __attribute__((ext_vector_type(8)))  __bf16 v8bf;
typedef __attribute__((ext_vector_type(8)))  float  v8f;
typedef __attribute__((ext_vector_type(4)))  float  v4f;
typedef __attribute__((ext_vector_type(4)))  unsigned int u32x4;
typedef __attribute__((ext_vector_type(8)))  unsigned int u32x8;

#define B_   8
#define T_   4096
#define D_   768
#define KT   32          // WMMA K-depth for bf16
#define NKK  (D_ / KT)   // 24 k-chunks along D

// ---------------------------------------------------------------------------
// Phase 0: f32 -> bf16 bulk convert (8 elements/thread, exact grids).
// ---------------------------------------------------------------------------
__global__ __launch_bounds__(256)
void cvt_kernel(const float* __restrict__ src, bf16_t* __restrict__ dst)
{
    const size_t i = ((size_t)blockIdx.x * 256 + threadIdx.x) * 8;
    v4f a = *(const v4f*)(src + i);
    v4f b = *(const v4f*)(src + i + 4);
    v8bf o;
#pragma unroll
    for (int j = 0; j < 4; ++j) { o[j] = (bf16_t)a[j]; o[4 + j] = (bf16_t)b[j]; }
    *(v8bf*)(dst + i) = o;
}

// ---------------------------------------------------------------------------
// Phase 1: one 16x16 output tile per wave. M = B*T rows, N = D cols.
// A lane mapping (16x32 bf16 A): row = lane%16, K = e + 8*((e>>3)+(lane>>4))
// B lane mapping (32x16 bf16 B): col = lane%16, K = e + 8*((e>>3)+(lane>>4))
//   Bmat[k][n] = W[n][k] -> per-lane contiguous along k in row-major W.
// ---------------------------------------------------------------------------
__global__ __launch_bounds__(128)
void proj_kernel(const bf16_t* __restrict__ Xb, const bf16_t* __restrict__ Wb,
                 const float* __restrict__ bias, bf16_t* __restrict__ Out)
{
    const int lane   = threadIdx.x & 31;
    const int wave   = threadIdx.x >> 5;
    const int tile   = blockIdx.x * 4 + wave;
    const int nTiles = D_ / 16;                 // 48
    const int mTile  = tile / nTiles;
    const int nTile  = tile % nTiles;
    const int col16  = lane & 15;
    const int half   = lane >> 4;

    const int mRow = mTile * 16 + col16;
    const int nCol = nTile * 16 + col16;

    v8f acc = {};
    for (int kk = 0; kk < NKK; ++kk) {
        const int kbase = kk * KT;

        const bf16_t* ap = Xb + (size_t)mRow * D_ + kbase + 8 * half;
        v8bf alo = *(const v8bf*)(ap);
        v8bf ahi = *(const v8bf*)(ap + 16);
        v16bf a;
#pragma unroll
        for (int i = 0; i < 8; ++i) { a[i] = alo[i]; a[8 + i] = ahi[i]; }

        const bf16_t* bp = Wb + (size_t)nCol * D_ + kbase + 8 * half;
        v8bf blo = *(const v8bf*)(bp);
        v8bf bhi = *(const v8bf*)(bp + 16);
        v16bf bb;
#pragma unroll
        for (int i = 0; i < 8; ++i) { bb[i] = blo[i]; bb[8 + i] = bhi[i]; }

        acc = __builtin_amdgcn_wmma_f32_16x16x32_bf16(
            false, a, false, bb, (short)0, acc, false, false);
    }

    const float bn = bias[nCol];
#pragma unroll
    for (int r = 0; r < 8; ++r) {
        const int m = mTile * 16 + r + 8 * half;
        Out[(size_t)m * D_ + nCol] = (bf16_t)(acc[r] + bn);
    }
}

// ---------------------------------------------------------------------------
// TDM: DMA one 16-row x 768-col bf16 K-tile (24 KB) from global into LDS.
// 2D descriptor, data_size = 8B units: row = 192 units, 16 rows, contiguous.
// D# group0 (4 SGPRs): count=1 | lds_addr | global_addr | type=2
// D# group1 (8 SGPRs): data_size=3, tensor_dim0=192, tensor_dim1=16,
//                      tile_dim0=192, tile_dim1=16, tensor_dim0_stride=192
// ---------------------------------------------------------------------------
__device__ __forceinline__ void tdm_load_ktile(const bf16_t* gsrc, unsigned lds_off)
{
    const unsigned long long ga = (unsigned long long)(uintptr_t)gsrc;
    u32x4 g0;
    g0[0] = 1u;                                            // count = 1 (valid)
    g0[1] = lds_off;                                       // lds_addr
    g0[2] = (unsigned)ga;                                  // global_addr[31:0]
    g0[3] = ((unsigned)(ga >> 32) & 0x1FFFFFFu) | (2u << 30); // ga[56:32] | type=2
    u32x8 g1;
    g1[0] = 0x00030000u;       // workgroup_mask=0, data_size=3 (8B), no pad/iter
    g1[1] = 192u << 16;        // tensor_dim0[15:0] in [31:16]
    g1[2] = 16u  << 16;        // tensor_dim1[15:0] in [31:16]
    g1[3] = 192u << 16;        // tile_dim0 in [31:16]
    g1[4] = 16u;               // tile_dim1 = 16, tile_dim2 = 0
    g1[5] = 192u;              // tensor_dim0_stride[31:0]
    g1[6] = 0u;                // stride hi, tensor_dim1_stride lo (2D: unused)
    g1[7] = 0u;
    asm volatile("tensor_load_to_lds %0, %1" :: "s"(g0), "s"(g1) : "memory");
}

// ---------------------------------------------------------------------------
// Phase 2: block = 4 waves, each wave owns one of 4 consecutive q-tiles for
// one batch. Q tiles staged per-wave in LDS as WMMA A-fragments; K tiles
// DMA'd by TDM into a double-buffered LDS strip shared by all 4 waves.
// ---------------------------------------------------------------------------
__global__ __launch_bounds__(128)
void attn_diag_kernel(const bf16_t* __restrict__ Qb, const bf16_t* __restrict__ Kb,
                      const int* __restrict__ mask, float* __restrict__ wdiag)
{
    __shared__ bf16_t qfrag[4][NKK][32][16];   // 96 KB: A-fragments per wave
    __shared__ bf16_t ktile[2][16][D_];        // 48 KB: double-buffered K tile

    const int lane  = threadIdx.x & 31;
    const int wave  = threadIdx.x >> 5;
    const int b     = blockIdx.x >> 6;               // /(T/16/4)
    const int qt    = ((blockIdx.x & 63) << 2) + wave;
    const int col16 = lane & 15;
    const int half  = lane >> 4;
    const size_t base = (size_t)b * T_ * D_;
    const int m0 = qt * 16;

    // Stage this wave's Q tile as WMMA A-fragments (per-lane private slots,
    // written and read by the same lane -> wave-internal DS ordering suffices).
    for (int kk = 0; kk < NKK; ++kk) {
        const bf16_t* qp = Qb + base + (size_t)(m0 + col16) * D_ + kk * KT + 8 * half;
        v8bf lo = *(const v8bf*)(qp);
        v8bf hi = *(const v8bf*)(qp + 16);
        *(v8bf*)&qfrag[wave][kk][lane][0] = lo;
        *(v8bf*)&qfrag[wave][kk][lane][8] = hi;
    }

    // Prime the TDM double buffer (wave 0 issues; TENSORcnt is per-wave).
    if (wave == 0) {
        tdm_load_ktile(Kb + base, (unsigned)(uintptr_t)&ktile[0][0][0]);
        tdm_load_ktile(Kb + base + (size_t)16 * D_, (unsigned)(uintptr_t)&ktile[1][0][0]);
    }

    const float scale = 0.03608439182435161f;   // 1/sqrt(768)
    float rowsum[8];
#pragma unroll
    for (int r = 0; r < 8; ++r) rowsum[r] = 0.0f;
    float pdiag = 0.0f;

    for (int kt = 0; kt < T_ / 16; ++kt) {
        const int buf = kt & 1;

        if (wave == 0) {
            if (kt < T_ / 16 - 1) __builtin_amdgcn_s_wait_tensorcnt(1);
            else                  __builtin_amdgcn_s_wait_tensorcnt(0);
        }
        __syncthreads();   // K tile kt is now visible to all waves

        const int nColG = kt * 16 + col16;
        v8f acc = {};
        for (int kk = 0; kk < NKK; ++kk) {
            v8bf alo = *(const v8bf*)&qfrag[wave][kk][lane][0];
            v8bf ahi = *(const v8bf*)&qfrag[wave][kk][lane][8];
            v16bf a;
#pragma unroll
            for (int i = 0; i < 8; ++i) { a[i] = alo[i]; a[8 + i] = ahi[i]; }

            const bf16_t* kp = &ktile[buf][col16][kk * KT + 8 * half];
            v8bf blo = *(const v8bf*)(kp);
            v8bf bhi = *(const v8bf*)(kp + 16);
            v16bf bb;
#pragma unroll
            for (int i = 0; i < 8; ++i) { bb[i] = blo[i]; bb[8 + i] = bhi[i]; }

            acc = __builtin_amdgcn_wmma_f32_16x16x32_bf16(
                false, a, false, bb, (short)0, acc, false, false);
        }

        const int mv = mask[b * T_ + nColG];    // same column for all 8 rows
#pragma unroll
        for (int r = 0; r < 8; ++r) {
            const float p = mv ? __expf(acc[r] * scale) : 0.0f;
            rowsum[r] += p;
            const int qrow = m0 + r + 8 * half;
            if (nColG == qrow) pdiag = p;       // diagonal element (kt == qt)
        }

        __syncthreads();   // everyone done reading buf before it is overwritten
        if (wave == 0 && kt + 2 < T_ / 16) {
            tdm_load_ktile(Kb + base + (size_t)((kt + 2) * 16) * D_,
                           (unsigned)(uintptr_t)&ktile[buf][0][0]);
        }
    }

    // Reduce each row's partial sums across the 16 lanes that share it.
#pragma unroll
    for (int r = 0; r < 8; ++r) {
#pragma unroll
        for (int off = 1; off < 16; off <<= 1)
            rowsum[r] += __shfl_xor(rowsum[r], off, 32);
    }

    // Diagonal holders: lanes 0..7 (rows m0+0..7), lanes 24..31 (rows m0+8..15)
    const int r = col16 - 8 * half;
    if (r >= 0 && r < 8) {
        wdiag[b * T_ + m0 + col16] = pdiag / rowsum[r];
    }
}

// ---------------------------------------------------------------------------
// Phase 3: out[b,d] = sum_t X[b,t,d] * wdiag[b,t]
// ---------------------------------------------------------------------------
__global__ void zero_out_kernel(float* __restrict__ out, int n)
{
    const int i = blockIdx.x * blockDim.x + threadIdx.x;
    if (i < n) out[i] = 0.0f;
}

__global__ __launch_bounds__(256)
void out_kernel(const float* __restrict__ X, const float* __restrict__ wdiag,
                float* __restrict__ out)
{
    const int chunks = T_ / 64;
    const int b  = blockIdx.x / chunks;
    const int tc = blockIdx.x % chunks;

    float acc[3] = {0.0f, 0.0f, 0.0f};
    for (int tt = 0; tt < 64; ++tt) {
        const int t = tc * 64 + tt;
        const float w = wdiag[b * T_ + t];
        const float* xp = X + ((size_t)b * T_ + t) * D_;
#pragma unroll
        for (int j = 0; j < 3; ++j)
            acc[j] += w * xp[threadIdx.x + 256 * j];
    }
#pragma unroll
    for (int j = 0; j < 3; ++j)
        atomicAdd(&out[b * D_ + threadIdx.x + 256 * j], acc[j]);
}

// ---------------------------------------------------------------------------
extern "C" void kernel_launch(void* const* d_in, const int* in_sizes, int n_in,
                              void* d_out, int out_size, void* d_ws, size_t ws_size,
                              hipStream_t stream)
{
    const float* X    = (const float*)d_in[0];
    const int*   mask = (const int*)d_in[1];
    const float* Wq   = (const float*)d_in[2];
    const float* bq   = (const float*)d_in[3];
    const float* Wk   = (const float*)d_in[4];
    const float* bk   = (const float*)d_in[5];
    float* out = (float*)d_out;

    const size_t NX = (size_t)B_ * T_ * D_;     // 25,165,824
    const size_t NW = (size_t)D_ * D_;          // 589,824

    bf16_t* Xb  = (bf16_t*)d_ws;
    bf16_t* Qb  = Xb + NX;
    bf16_t* Kb  = Qb + NX;
    bf16_t* Wqb = Kb + NX;
    bf16_t* Wkb = Wqb + NW;
    float*  wd  = (float*)(Wkb + NW);

    // Phase 0: bf16 staging (exact grids: sizes divisible by 2048)
    cvt_kernel<<<NX / 2048, 256, 0, stream>>>(X,  Xb);
    cvt_kernel<<<NW / 2048, 256, 0, stream>>>(Wq, Wqb);
    cvt_kernel<<<NW / 2048, 256, 0, stream>>>(Wk, Wkb);

    // Phase 1: projections
    const int tilesM = (B_ * T_) / 16;          // 2048
    const int tilesN = D_ / 16;                 // 48
    proj_kernel<<<tilesM * tilesN / 4, 128, 0, stream>>>(Xb, Wqb, bq, Qb);
    proj_kernel<<<tilesM * tilesN / 4, 128, 0, stream>>>(Xb, Wkb, bk, Kb);

    // Phase 2: masked-softmax diagonal weights
    attn_diag_kernel<<<B_ * (T_ / 16) / 4, 128, 0, stream>>>(Qb, Kb, mask, wd);

    // Phase 3: weighted reduction over T
    zero_out_kernel<<<(B_ * D_ + 255) / 256, 256, 0, stream>>>(out, B_ * D_);
    out_kernel<<<B_ * (T_ / 64), 256, 0, stream>>>(X, wd, out);
}
